// Model_31430570672210
// MI455X (gfx1250) — matrix operand order repaired and checked by reading, not compiled, not run on hardware
//
#include <hip/hip_runtime.h>

// ---------------------------------------------------------------------------
// CDNA5 (gfx1250), wave32 + V_WMMA_F32_16X16X32_BF16 + async-to-LDS staging.
// Shapes fixed: B=32, T=1024, W=768, N=256, K=4096, D=64.
//
// ~31.7 GFLOP of GEMM vs ~250MB HBM traffic (memory floor ~11us @ 23.3TB/s).
// All B operands (and transposed-A operands) are packed ONCE into WMMA
// fragment-major bf16 (each lane's 16 fragment elements contiguous, 32B).
// The two large GEMMs use 32x64 register blocks (8 WMMA per K-step from
// 2 A + 4 B fragments; the 4 B tiles are contiguous 4KB) and stage the
// workgroup-shared A block into LDS with global_load_async_to_lds_b128
// (ASYNCcnt + barrier), so A is fetched once per workgroup, not per wave.
// Codewords are L2-normalized (||e||=1) so cdist+argmin collapses to argmax
// over z2n @ e^T, fused with a branchless per-lane running argmax.
// ---------------------------------------------------------------------------

typedef __attribute__((ext_vector_type(16))) __bf16 v16bf;
typedef __attribute__((ext_vector_type(8)))  __bf16 v8bfh;
typedef __attribute__((ext_vector_type(8)))  float  v8f;

#define WMMA_BF16(A, Bf, Cacc) \
  __builtin_amdgcn_wmma_f32_16x16x32_bf16(false, (A), false, (Bf), (short)0, (Cacc), false, false)

// Load one fragment from fragment-major packed storage: 2 x 16B contiguous.
__device__ __forceinline__ v16bf load_pk(const __bf16* __restrict__ p) {
  v8bfh lo = *(const v8bfh*)(p);
  v8bfh hi = *(const v8bfh*)(p + 8);
  v16bf f;
#pragma unroll
  for (int e = 0; e < 8; ++e) { f[e] = lo[e]; f[e + 8] = hi[e]; }
  return f;
}

// A-fragment (row-major bf16 source): 16x32, ISA 7.12.2 layout.
template <int LDA>
__device__ __forceinline__ v16bf load_a_n(const __bf16* __restrict__ A, int row0, int k0) {
  const int lane = threadIdx.x & 31;
  const int h16  = lane >> 4;
  const int lid  = lane & 15;
  const __bf16* p = A + (row0 + lid) * LDA + k0 + 8 * h16;
  v8bfh lo = *(const v8bfh*)(p);
  v8bfh hi = *(const v8bfh*)(p + 16);
  v16bf f;
#pragma unroll
  for (int e = 0; e < 8; ++e) { f[e] = lo[e]; f[e + 8] = hi[e]; }
  return f;
}

// Pack B[k,n] (or B=src^T) tiles into fragment-major bf16.
template <typename SRC, int SRCLD, bool SRCT, int NT>
__global__ __launch_bounds__(256) void pack_b(const SRC* __restrict__ src,
                                              __bf16* __restrict__ dst,
                                              long srcStride, long dstStride) {
  const int wave = threadIdx.x >> 5, lane = threadIdx.x & 31;
  const int h16 = lane >> 4, lid = lane & 15;
  int nt = blockIdx.x * (blockDim.x >> 5) + wave;
  if (nt >= NT) return;
  int kt = blockIdx.y, bz = blockIdx.z;
  const SRC* s = src + (long)bz * srcStride;
  __bf16* d = dst + (long)bz * dstStride + (long)(kt * NT + nt) * 512 + lane * 16;
  int n = nt * 16 + lid, kb = kt * 32 + 8 * h16;
#pragma unroll
  for (int e = 0; e < 16; ++e) {
    int ke = (e < 8) ? e : (e + 8);
    float v = SRCT ? (float)s[n * SRCLD + kb + ke] : (float)s[(kb + ke) * SRCLD + n];
    d[e] = (__bf16)v;
  }
}

// Pack transposed-A operands (A(m,k) = src[k*SRCLD + m]) into fragment-major.
template <typename SRC, int SRCLD, int MT, int KT>
__global__ __launch_bounds__(256) void pack_a(const SRC* __restrict__ src,
                                              __bf16* __restrict__ dst) {
  const int wave = threadIdx.x >> 5, lane = threadIdx.x & 31;
  const int h16 = lane >> 4, lid = lane & 15;
  int mt = blockIdx.x * (blockDim.x >> 5) + wave;
  if (mt >= MT) return;
  int kt = blockIdx.y;
  __bf16* d = dst + (long)(mt * KT + kt) * 512 + lane * 16;
  int m = mt * 16 + lid;
#pragma unroll
  for (int e = 0; e < 16; ++e) {
    int keff = (e < 8) ? (e + 8 * h16) : (e + 8 + 8 * h16);
    d[e] = (__bf16)(float)src[(kt * 32 + keff) * SRCLD + m];
  }
}

// -------- big blocked GEMM: packed A (LDS-staged) x packed B ---------------
// Wave computes a 32x64 C block (2 m-tiles x 4 n-tiles, 8 WMMA per K-step).
// The workgroup's A block (2 m-tiles x KT k-tiles, contiguous in packed
// layout) is staged into LDS once via async-to-LDS; all 4 waves share it.
template <int NB, int KT, int LDC, bool OUTBF>   // NB = #64-col n-blocks
__global__ __launch_bounds__(128) void gemm_blk(
    const __bf16* __restrict__ Apk, const __bf16* __restrict__ Bp,
    void* __restrict__ C, long bStride, long cStride) {
  __shared__ __bf16 As[2 * KT * 512];
  const int wave = threadIdx.x >> 5;
  const int lane = threadIdx.x & 31;
  const int h16  = lane >> 4;
  const int lid  = lane & 15;
  const int mb = blockIdx.y;
  const int bz = blockIdx.z;

  // Stage A block (2*KT*1024 bytes, contiguous) into LDS: each thread issues
  // 16B async copies; wait ASYNCcnt then barrier.
  {
    const char* g = (const char*)(Apk + (long)(2 * mb) * KT * 512);
    constexpr int bytes = 2 * KT * 512 * 2;
#pragma unroll
    for (int off = 0; off < bytes; off += 128 * 16) {
      int o = off + threadIdx.x * 16;
      unsigned l = (unsigned)(unsigned long long)(uintptr_t)((char*)As + o);
      const char* gp = g + o;
      asm volatile("global_load_async_to_lds_b128 %0, %1, off"
                   :: "v"(l), "v"(gp) : "memory");
    }
    asm volatile("s_wait_asynccnt 0" ::: "memory");
    __syncthreads();
  }

  const int nb = blockIdx.x * 4 + wave;
  if (nb >= NB) return;
  const __bf16* pb = Bp + (long)bz * bStride + (long)nb * 4 * 512 + lane * 16;
  v8f acc[2][4] = {};
  for (int kt = 0; kt < KT; ++kt) {
    v16bf a0 = load_pk(As + kt * 512 + lane * 16);          // m-tile 0 (ds_load)
    v16bf a1 = load_pk(As + (KT + kt) * 512 + lane * 16);   // m-tile 1
    v16bf b0 = load_pk(pb);
    v16bf b1 = load_pk(pb + 512);
    v16bf b2 = load_pk(pb + 1024);
    v16bf b3 = load_pk(pb + 1536);
    pb += (long)NB * 4 * 512;
    acc[0][0] = WMMA_BF16(a0, b0, acc[0][0]);
    acc[0][1] = WMMA_BF16(a0, b1, acc[0][1]);
    acc[0][2] = WMMA_BF16(a0, b2, acc[0][2]);
    acc[0][3] = WMMA_BF16(a0, b3, acc[0][3]);
    acc[1][0] = WMMA_BF16(a1, b0, acc[1][0]);
    acc[1][1] = WMMA_BF16(a1, b1, acc[1][1]);
    acc[1][2] = WMMA_BF16(a1, b2, acc[1][2]);
    acc[1][3] = WMMA_BF16(a1, b3, acc[1][3]);
  }
#pragma unroll
  for (int mi = 0; mi < 2; ++mi)
#pragma unroll
    for (int j = 0; j < 4; ++j)
#pragma unroll
      for (int v = 0; v < 8; ++v) {
        int row = mb * 32 + mi * 16 + v + 8 * h16;
        int col = (nb * 4 + j) * 16 + lid;
        float r = acc[mi][j][v];
        if (OUTBF) ((__bf16*)C + (long)bz * cStride)[row * LDC + col] = (__bf16)r;
        else       ((float*)C  + (long)bz * cStride)[row * LDC + col] = r;
      }
}

// -------- small GEMM (wave per 16x16 tile): row-major A x packed B ---------
template <int LDA, int NT, int LDC, bool OUTBF>
__global__ __launch_bounds__(256) void gemm_sm(
    const __bf16* __restrict__ A, const __bf16* __restrict__ Bp,
    void* __restrict__ C, const float* __restrict__ bias, int KT) {
  const int wave = threadIdx.x >> 5;
  const int lane = threadIdx.x & 31;
  const int h16  = lane >> 4;
  const int lid  = lane & 15;
  const int nt = blockIdx.x * (blockDim.x >> 5) + wave;
  if (nt >= NT) return;
  const int mt = blockIdx.y;
  const __bf16* pb = Bp + (long)nt * 512 + lane * 16;
  v8f acc = {};
  for (int kt = 0; kt < KT; ++kt) {
    v16bf af = load_a_n<LDA>(A, mt * 16, kt * 32);
    v16bf bf = load_pk(pb);
    pb += (long)NT * 512;
    acc = WMMA_BF16(af, bf, acc);
  }
  const int col = nt * 16 + lid;
  const float bv = bias ? bias[col] : 0.0f;
#pragma unroll
  for (int v = 0; v < 8; ++v) {
    int row = mt * 16 + v + 8 * h16;
    float r = acc[v] + bv;
    if (OUTBF) ((__bf16*)C)[row * LDC + col] = (__bf16)r;
    else       ((float*)C)[row * LDC + col] = r;
  }
}

// jax.nn.standardize(axis=-1) over rows of 64; one wave per row.
__global__ __launch_bounds__(256) void standardize_rows64(
    float* __restrict__ z, __bf16* __restrict__ zbf, int rows) {
  int row = blockIdx.x * (blockDim.x >> 5) + (threadIdx.x >> 5);
  if (row >= rows) return;
  int lane = threadIdx.x & 31;
  float* p = z + row * 64;
  float a = p[lane], b = p[lane + 32];
  float s = a + b, q = a * a + b * b;
#pragma unroll
  for (int m = 16; m >= 1; m >>= 1) {
    s += __shfl_xor(s, m, 32);
    q += __shfl_xor(q, m, 32);
  }
  float mean = s * (1.0f / 64.0f);
  float var  = q * (1.0f / 64.0f) - mean * mean;
  float r = rsqrtf(var + 1e-5f);
  a = (a - mean) * r;
  b = (b - mean) * r;
  p[lane]      = a;
  p[lane + 32] = b;
  zbf[row * 64 + lane]      = (__bf16)a;
  zbf[row * 64 + lane + 32] = (__bf16)b;
}

// e = L2norm(dict / counts), f32.
__global__ __launch_bounds__(256) void codewords(
    const float* __restrict__ dict, const float* __restrict__ cnt,
    float* __restrict__ e, int K) {
  int row = blockIdx.x * (blockDim.x >> 5) + (threadIdx.x >> 5);
  if (row >= K) return;
  int lane = threadIdx.x & 31;
  float inv = 1.0f / cnt[row];
  float a = dict[row * 64 + lane] * inv;
  float b = dict[row * 64 + lane + 32] * inv;
  float q = a * a + b * b;
#pragma unroll
  for (int m = 16; m >= 1; m >>= 1) q += __shfl_xor(q, m, 32);
  float r = rsqrtf(q);
  e[row * 64 + lane]      = a * r;
  e[row * 64 + lane + 32] = b * r;
}

// Fused scores GEMM + argmax over 32 rows per wave (4 WMMA per B pair, B
// fragment-packed and shared by both row-tiles).  argmax == argmin distance
// since ||e||=1 and ||z||^2 is constant per row; ties -> lowest index.
__global__ __launch_bounds__(256) void vq_argmax(
    const __bf16* __restrict__ znbf, const __bf16* __restrict__ eTp,
    int rows, int* __restrict__ idx_i, float* __restrict__ idx_f) {
  constexpr int NT = 256;                    // K/16 column tiles
  int rt = blockIdx.x * (blockDim.x >> 5) + (threadIdx.x >> 5);
  if (rt * 32 >= rows) return;
  int lane = threadIdx.x & 31;
  int h16 = lane >> 4, lid = lane & 15;
  int row0 = rt * 32;
  v16bf a00 = load_a_n<64>(znbf, row0, 0);
  v16bf a01 = load_a_n<64>(znbf, row0, 32);
  v16bf a10 = load_a_n<64>(znbf, row0 + 16, 0);
  v16bf a11 = load_a_n<64>(znbf, row0 + 16, 32);
  float bv0[8], bv1[8]; int bi0[8], bi1[8];
#pragma unroll
  for (int v = 0; v < 8; ++v) {
    bv0[v] = -3.4e38f; bi0[v] = 0;
    bv1[v] = -3.4e38f; bi1[v] = 0;
  }
  const __bf16* pb = eTp + lane * 16;
  for (int nt = 0; nt < NT; ++nt) {
    v16bf b0 = load_pk(pb + (long)nt * 512);
    v16bf b1 = load_pk(pb + (long)(NT + nt) * 512);
    v8f c0 = {}, c1 = {};
    c0 = WMMA_BF16(a00, b0, c0);
    c0 = WMMA_BF16(a01, b1, c0);
    c1 = WMMA_BF16(a10, b0, c1);
    c1 = WMMA_BF16(a11, b1, c1);
    int ix = nt * 16 + lid;
#pragma unroll
    for (int v = 0; v < 8; ++v) {
      bool t0 = c0[v] > bv0[v];
      bv0[v] = t0 ? c0[v] : bv0[v];
      bi0[v] = t0 ? ix : bi0[v];
      bool t1 = c1[v] > bv1[v];
      bv1[v] = t1 ? c1[v] : bv1[v];
      bi1[v] = t1 ? ix : bi1[v];
    }
  }
#pragma unroll
  for (int half = 0; half < 2; ++half) {
#pragma unroll
    for (int v = 0; v < 8; ++v) {
      float val = half ? bv1[v] : bv0[v];
      int   ix  = half ? bi1[v] : bi0[v];
#pragma unroll
      for (int s = 8; s >= 1; s >>= 1) {     // butterfly within 16-lane half
        float ov = __shfl_xor(val, s, 32);
        int   oi = __shfl_xor(ix, s, 32);
        bool t = (ov > val) || (ov == val && oi < ix);
        val = t ? ov : val;
        ix  = t ? oi : ix;
      }
      if (lid == 0) {
        int row = row0 + half * 16 + v + 8 * h16;
        idx_i[row] = ix;
        idx_f[row] = (float)ix;
      }
    }
  }
}

// zq = e[idx] (f32 out + bf16 for dec GEMM); counts histogram; x_sum scatter.
__global__ __launch_bounds__(256) void vq_apply(
    const int* __restrict__ idx, const float* __restrict__ e,
    const float* __restrict__ zn, float* __restrict__ zq, __bf16* __restrict__ zqbf,
    float* __restrict__ cnt, float* __restrict__ xsum, int rows) {
  int row = blockIdx.x * (blockDim.x >> 5) + (threadIdx.x >> 5);
  if (row >= rows) return;
  int lane = threadIdx.x & 31;
  int k = idx[row];
  float e0 = e[k * 64 + lane], e1 = e[k * 64 + lane + 32];
  zq[row * 64 + lane]        = e0;
  zq[row * 64 + lane + 32]   = e1;
  zqbf[row * 64 + lane]      = (__bf16)e0;
  zqbf[row * 64 + lane + 32] = (__bf16)e1;
  atomicAdd(&xsum[k * 64 + lane],      zn[row * 64 + lane]);
  atomicAdd(&xsum[k * 64 + lane + 32], zn[row * 64 + lane + 32]);
  if (lane == 0) atomicAdd(&cnt[k], 1.0f);
}

__global__ __launch_bounds__(256) void zero_f32(float* __restrict__ p, int n) {
  int i = blockIdx.x * blockDim.x + threadIdx.x;
  if (i < n) p[i] = 0.0f;
}

extern "C" void kernel_launch(void* const* d_in, const int* in_sizes, int n_in,
                              void* d_out, int out_size, void* d_ws, size_t ws_size,
                              hipStream_t stream) {
  const float* x    = (const float*)d_in[0];   // [B,T,W]
  const float* ds   = (const float*)d_in[1];   // [T,N]
  const float* ehw  = (const float*)d_in[2];   // [W,D]
  const float* ehb  = (const float*)d_in[3];   // [D]
  const float* dict = (const float*)d_in[4];   // [K,D]
  const float* cnt  = (const float*)d_in[5];   // [K]
  const float* dsw  = (const float*)d_in[6];   // [D,W]
  const float* dsb  = (const float*)d_in[7];   // [W]
  const float* up   = (const float*)d_in[8];   // [N,T]

  const int B = 32, T = 1024, W = 768, N = 256, K = 4096, D = 64;
  const int rows = B * N;                      // 8192

  float* out     = (float*)d_out;
  float* y_out   = out;                        // B*T*W
  float* zq_out  = y_out + (long)B * T * W;    // rows*D
  float* idx_out = zq_out + (long)rows * D;    // rows
  float* cnt_out = idx_out + rows;             // K
  float* xs_out  = cnt_out + K;                // K*D

  char*  w   = (char*)d_ws;
  size_t off = 0;
  auto carve = [&](size_t bytes) {
    void* p = w + off;
    off = (off + bytes + 255) & ~(size_t)255;
    return p;
  };
  __bf16* xpk    = (__bf16*)carve((size_t)B * T * W * 2);   // x packed (B of g1)
  __bf16* dspk   = (__bf16*)carve((size_t)T * N * 2);       // ds^T packed (A of g1)
  __bf16* ehwpk  = (__bf16*)carve((size_t)W * D * 2);       // ehw packed (B of g2)
  __bf16* dswpk  = (__bf16*)carve((size_t)D * W * 2);       // dsw packed (B of g3)
  __bf16* uppk   = (__bf16*)carve((size_t)N * T * 2);       // up^T packed (A of g4)
  __bf16* z1bf   = (__bf16*)carve((size_t)rows * W * 2);    // stage-1 out, row-major
  float*  zn     = (float*)carve((size_t)rows * D * 4);     // standardized z (f32)
  __bf16* znbf   = (__bf16*)carve((size_t)rows * D * 2);    // bf16 copy for argmax
  float*  e      = (float*)carve((size_t)K * D * 4);        // codewords f32
  __bf16* eTpk   = (__bf16*)carve((size_t)D * K * 2);       // e^T packed (B of scores)
  __bf16* zqbf   = (__bf16*)carve((size_t)rows * D * 2);    // quantized z, row-major
  __bf16* dbufbf = (__bf16*)carve((size_t)rows * W * 2);    // decoded stem, row-major
  __bf16* dbufpk = (__bf16*)carve((size_t)rows * W * 2);    // decoded stem packed
  int*    idxw   = (int*)carve((size_t)rows * 4);

  // Zero accumulated output regions (counts_new + x_sum) every launch.
  {
    int nz = K + K * D;
    zero_f32<<<dim3((nz + 255) / 256), dim3(256), 0, stream>>>(cnt_out, nz);
  }

  // --- pack passes (strided gather + cvt exactly once per element) ---
  pack_b<float, 768, false, 48><<<dim3(6, 32, 32), dim3(256), 0, stream>>>(
      x, xpk, (long)T * W, (long)T * W);
  pack_a<float, 256, 16, 32><<<dim3(2, 32, 1), dim3(256), 0, stream>>>(ds, dspk);
  pack_b<float, 64, false, 4><<<dim3(1, 24, 1), dim3(128), 0, stream>>>(
      ehw, ehwpk, 0, 0);
  pack_b<float, 768, false, 48><<<dim3(6, 2, 1), dim3(256), 0, stream>>>(
      dsw, dswpk, 0, 0);
  pack_a<float, 1024, 64, 8><<<dim3(8, 8, 1), dim3(256), 0, stream>>>(up, uppk);

  // g1: z1[b] = ds^T @ x[b] : M=256, N=768, Kdim=1024; bf16 row-major out.
  // 12 n-blocks of 64 cols, 8 m-blocks of 32 rows, LDS-staged A (64KB).
  gemm_blk<12, 32, 768, true><<<dim3(3, 8, 32), dim3(128), 0, stream>>>(
      dspk, xpk, z1bf, (long)T * W, (long)N * W);

  // g2: zn = z1 @ enc_head_w + b : M=rows, N=64, Kdim=768; f32 out.
  gemm_sm<768, 4, 64, false><<<dim3(1, rows / 16, 1), dim3(128), 0, stream>>>(
      z1bf, ehwpk, zn, ehb, 24);

  standardize_rows64<<<dim3(rows / 8), dim3(256), 0, stream>>>(zn, znbf, rows);

  codewords<<<dim3(K / 8), dim3(256), 0, stream>>>(dict, cnt, e, K);

  // e^T as B (coalesced gather from e): N=K(256 tiles), K=D(2 tiles).
  pack_b<float, 64, true, 256><<<dim3(32, 2, 1), dim3(256), 0, stream>>>(
      e, eTpk, 0, 0);

  vq_argmax<<<dim3(rows / 32 / 8), dim3(256), 0, stream>>>(znbf, eTpk, rows, idxw, idx_out);

  vq_apply<<<dim3(rows / 8), dim3(256), 0, stream>>>(
      idxw, e, zn, zq_out, zqbf, cnt_out, xs_out, rows);

  // g3: dbuf = zq @ dec_stem_w + b : M=rows, N=768, Kdim=64; bf16 out.
  gemm_sm<64, 48, 768, true><<<dim3(6, rows / 16, 1), dim3(256), 0, stream>>>(
      zqbf, dswpk, dbufbf, dsb, 2);

  // dbuf[b] as B for g4: N=W(48 tiles), K=N(8 tiles), per batch.
  pack_b<__bf16, 768, false, 48><<<dim3(6, 8, 32), dim3(256), 0, stream>>>(
      dbufbf, dbufpk, (long)N * W, (long)N * W);

  // g4: y[b] = up^T @ dbuf[b] : M=1024, N=768, Kdim=256; f32 out.
  // 12 n-blocks, 32 m-blocks, LDS-staged A (16KB).
  gemm_blk<12, 8, 768, false><<<dim3(3, 32, 32), dim3(128), 0, stream>>>(
      uppk, dbufpk, y_out, (long)N * W, (long)T * W);
}
